// MixtralOfExpertsLayer_75797582840348
// MI455X (gfx1250) — matrix operand order repaired and checked by reading, not compiled, not run on hardware
//
#include <hip/hip_runtime.h>

// ---------------- problem constants ----------------
#define N_TOK 8192   // B*T
#define DIM_D 1024
#define DIM_H 2048
#define DIM_O 1024
#define NEXP  8

// ---------------- GEMM tiling ----------------
#define BM 128
#define BN 256
#define BK 32
#define LDS_STRIDE 40   // 32 + 8 halfword pad; 80B row pitch keeps 16B alignment

typedef __attribute__((ext_vector_type(16))) __bf16 v16bf;
typedef __attribute__((ext_vector_type(8)))  float  v8f;

struct alignas(16) Frag32B { uint4 lo, hi; };

__device__ __forceinline__ unsigned short f2bf(float f) {
    unsigned int u = __builtin_bit_cast(unsigned int, f);
    u += 0x7FFFu + ((u >> 16) & 1u);           // round-to-nearest-even
    return (unsigned short)(u >> 16);
}

// A fragment (16x32 bf16): lane m=lane&15; VGPR0..3 <- K = 8*(lane>>4)..+7,
// VGPR4..7 <- K = 16 + 8*(lane>>4)..+7   (two 16B LDS loads)
__device__ __forceinline__ v16bf load_fragA(const unsigned short* s, int row_base, int lane) {
    const unsigned short* p = s + (size_t)(row_base + (lane & 15)) * LDS_STRIDE;
    const int kb = (lane >> 4) << 3;
    Frag32B f;
    f.lo = *(const uint4*)(p + kb);
    f.hi = *(const uint4*)(p + 16 + kb);
    return __builtin_bit_cast(v16bf, f);
}

// B fragment (32x16 bf16, operand pre-transposed so row n is K-contiguous):
// lane n=lane&15; VGPR0..3 <- K = 8*(lane>>4)..+7, VGPR4..7 <- K = 16 + 8*(lane>>4)..+7 (two 16B LDS loads)
__device__ __forceinline__ v16bf load_fragB(const unsigned short* s, int row_base, int lane) {
    const unsigned short* p = s + (size_t)(row_base + (lane & 15)) * LDS_STRIDE
                                + ((lane >> 4) << 3);
    Frag32B f;
    f.lo = *(const uint4*)(p);
    f.hi = *(const uint4*)(p + 16);
    return __builtin_bit_cast(v16bf, f);
}

// Async global->LDS staging of a rows x 32 bf16 tile (CDNA5 ASYNCcnt path, no VGPR data)
__device__ __forceinline__ void async_copy_tile(
    const unsigned short* __restrict__ gsrc, int ld, int row0, int k0,
    unsigned short* sdst, int rows, int tid)
{
    const int total = rows * 4;                 // 16B chunks in the tile
    for (int c = tid; c < total; c += 256) {
        const int row = c >> 2;
        const int kc  = (c & 3) << 3;           // halfword offset in 32-wide row
        const unsigned long long ga =
            (unsigned long long)(const void*)(gsrc + (size_t)(row0 + row) * ld + k0 + kc);
        const unsigned la =
            (unsigned)(unsigned long long)(void*)(sdst + row * LDS_STRIDE + kc);
        asm volatile("global_load_async_to_lds_b128 %0, %1, off"
                     :: "v"(la), "v"(ga) : "memory");
    }
}

__device__ __forceinline__ void wait_async0() {
    asm volatile("s_wait_asynccnt 0x0" ::: "memory");
}

// ---------------- gating: softmax(x@Wg+bg) -> top-2 -> L1 renorm ----------------
__global__ __launch_bounds__(256) void gate_kernel(
    const float* __restrict__ x, const float* __restrict__ Wg,
    const float* __restrict__ bg, float* __restrict__ g)
{
    const int t = blockIdx.x * blockDim.x + threadIdx.x;
    if (t >= N_TOK) return;
    float l[NEXP];
#pragma unroll
    for (int e = 0; e < NEXP; ++e) l[e] = bg[e];
    const float* xr = x + (size_t)t * DIM_D;
    for (int d = 0; d < DIM_D; ++d) {
        const float xv = xr[d];
        const float4 w0 = ((const float4*)(Wg + d * NEXP))[0];
        const float4 w1 = ((const float4*)(Wg + d * NEXP))[1];
        l[0] += xv * w0.x; l[1] += xv * w0.y; l[2] += xv * w0.z; l[3] += xv * w0.w;
        l[4] += xv * w1.x; l[5] += xv * w1.y; l[6] += xv * w1.z; l[7] += xv * w1.w;
    }
    float mx = l[0];
#pragma unroll
    for (int e = 1; e < NEXP; ++e) mx = fmaxf(mx, l[e]);
    float p[NEXP], s = 0.f;
#pragma unroll
    for (int e = 0; e < NEXP; ++e) { p[e] = __expf(l[e] - mx); s += p[e]; }
    const float inv = 1.f / s;
#pragma unroll
    for (int e = 0; e < NEXP; ++e) p[e] *= inv;
    int i1 = 0;
#pragma unroll
    for (int e = 1; e < NEXP; ++e) if (p[e] > p[i1]) i1 = e;
    int i2 = (i1 == 0) ? 1 : 0;
#pragma unroll
    for (int e = 0; e < NEXP; ++e) if (e != i1 && p[e] > p[i2]) i2 = e;
    const float denom = fmaxf(p[i1] + p[i2], 1e-12f);
#pragma unroll
    for (int e = 0; e < NEXP; ++e)
        g[(size_t)t * NEXP + e] = (e == i1 || e == i2) ? p[e] / denom : 0.f;
}

// ---------------- f32 -> bf16 elementwise (float4 granularity) ----------------
__global__ __launch_bounds__(256) void cvt_f32_bf16(
    const float* __restrict__ in, unsigned short* __restrict__ out, int n4)
{
    const int i = blockIdx.x * blockDim.x + threadIdx.x;
    if (i >= n4) return;
    const float4 f = ((const float4*)in)[i];
    uint2 r;
    r.x = (unsigned)f2bf(f.x) | ((unsigned)f2bf(f.y) << 16);
    r.y = (unsigned)f2bf(f.z) | ((unsigned)f2bf(f.w) << 16);
    ((uint2*)out)[i] = r;
}

// ---------------- tiled transpose + convert: in[E][R][C] f32 -> out[E][C][R] bf16 ----------------
__global__ __launch_bounds__(256) void transpose_cvt(
    const float* __restrict__ in, unsigned short* __restrict__ out, int R, int C)
{
    __shared__ unsigned short tile[32][33];
    const size_t base = (size_t)blockIdx.z * R * C;
    const int c0 = blockIdx.x * 32, r0 = blockIdx.y * 32;
    const int tx = threadIdx.x & 31, ty = threadIdx.x >> 5;
#pragma unroll
    for (int i = 0; i < 4; ++i) {
        const int r = ty + i * 8;
        tile[r][tx] = f2bf(in[base + (size_t)(r0 + r) * C + (c0 + tx)]);
    }
    __syncthreads();
#pragma unroll
    for (int i = 0; i < 4; ++i) {
        const int r = ty + i * 8;
        out[base + (size_t)(c0 + r) * R + (r0 + tx)] = tile[tx][r];
    }
}

__global__ __launch_bounds__(256) void zero_f32(float* __restrict__ p, int n)
{
    const int i = blockIdx.x * blockDim.x + threadIdx.x;
    if (i < n) p[i] = 0.f;
}

// ---------------- bf16 WMMA GEMM:  C[M,N] = A[M,K] @ Bt[N,K]^T (+bias) ----------------
// Block tile 128x256, 8 waves (2x4), each wave 64x64 (4x4 accumulators of 16x16).
// Double-buffered LDS staged with global_load_async_to_lds_b128 (ASYNCcnt).
// MODE 0: out bf16 = relu(acc + bias)            (expert up-proj -> hb)
// MODE 1: outF[m,n] += g[m,e] * (acc + bias)     (expert down-proj, weighted combine)
template <int MODE>
__global__ __launch_bounds__(256) void gemm_bf16_wmma(
    const unsigned short* __restrict__ A,
    const unsigned short* __restrict__ Bt,
    const float* __restrict__ bias,
    const float* __restrict__ gate, int expert,
    unsigned short* __restrict__ outB,
    float* __restrict__ outF,
    int M, int Ncols, int K)
{
    __shared__ __align__(16) unsigned short sA[2][BM * LDS_STRIDE];  // 2 x 10240B
    __shared__ __align__(16) unsigned short sB[2][BN * LDS_STRIDE];  // 2 x 20480B

    const int bm = blockIdx.y * BM;
    const int bn = blockIdx.x * BN;
    const int tid  = threadIdx.x;
    const int lane = tid & 31;
    const int wid  = tid >> 5;          // 8 waves: 2 (M) x 4 (N)
    const int wm = (wid >> 2) * 64;
    const int wn = (wid & 3) * 64;

    v8f acc[4][4];
#pragma unroll
    for (int i = 0; i < 4; ++i)
#pragma unroll
        for (int j = 0; j < 4; ++j)
            acc[i][j] = (v8f){0.f, 0.f, 0.f, 0.f, 0.f, 0.f, 0.f, 0.f};

    // prologue: stage tile 0
    async_copy_tile(A,  K, bm, 0, sA[0], BM, tid);
    async_copy_tile(Bt, K, bn, 0, sB[0], BN, tid);
    wait_async0();
    __syncthreads();

    int buf = 0;
    for (int k0 = 0; k0 < K; k0 += BK) {
        const int nxt = buf ^ 1;
        if (k0 + BK < K) {              // stage next tile while computing this one
            async_copy_tile(A,  K, bm, k0 + BK, sA[nxt], BM, tid);
            async_copy_tile(Bt, K, bn, k0 + BK, sB[nxt], BN, tid);
        }
        if (k0 + 2 * BK < K) {          // warm L2 two tiles ahead
            __builtin_prefetch(A  + (size_t)(bm + (tid >> 1)) * K + k0 + 2 * BK
                                  + ((tid & 1) << 4), 0, 1);
            __builtin_prefetch(Bt + (size_t)(bn + tid) * K + k0 + 2 * BK, 0, 1);
        }

        v16bf af[4], bfr[4];
#pragma unroll
        for (int i = 0; i < 4; ++i) af[i]  = load_fragA(sA[buf], wm + i * 16, lane);
#pragma unroll
        for (int j = 0; j < 4; ++j) bfr[j] = load_fragB(sB[buf], wn + j * 16, lane);

#pragma unroll
        for (int i = 0; i < 4; ++i)
#pragma unroll
            for (int j = 0; j < 4; ++j)
                acc[i][j] = __builtin_amdgcn_wmma_f32_16x16x32_bf16(
                    false, af[i], false, bfr[j], (short)0, acc[i][j], false, false);

        wait_async0();                  // my next-tile async copies landed
        __syncthreads();                // everyone done reading buf + all copies visible
        buf = nxt;
    }

    // epilogue: C VGPR v holds M = v + 8*(lane>>4), N = lane&15 (per 16x16 subtile)
    const int nl = lane & 15;
    const int mh = (lane >> 4) << 3;
#pragma unroll
    for (int i = 0; i < 4; ++i) {
#pragma unroll
        for (int j = 0; j < 4; ++j) {
            const int col   = bn + wn + j * 16 + nl;
            const int mbase = bm + wm + i * 16 + mh;
            const float bv  = bias[col];
#pragma unroll
            for (int v = 0; v < 8; ++v) {
                const int m = mbase + v;
                float val = acc[i][j][v] + bv;
                if (MODE == 0) {
                    val = val > 0.f ? val : 0.f;
                    outB[(size_t)m * Ncols + col] = f2bf(val);
                } else {
                    const float gg = gate[(size_t)m * NEXP + expert];
                    outF[(size_t)m * Ncols + col] += gg * val;
                }
            }
        }
    }
}

// ---------------- host-side launch ----------------
extern "C" void kernel_launch(void* const* d_in, const int* in_sizes, int n_in,
                              void* d_out, int out_size, void* d_ws, size_t ws_size,
                              hipStream_t stream) {
    const float* x  = (const float*)d_in[0];
    // d_in[1] = num_experts_chosen (== 2, hardcoded in gate_kernel)
    const float* Wg = (const float*)d_in[2];
    const float* bg = (const float*)d_in[3];
    const float* W1 = (const float*)d_in[4];
    const float* b1 = (const float*)d_in[5];
    const float* W2 = (const float*)d_in[6];
    const float* b2 = (const float*)d_in[7];
    float* out = (float*)d_out;

    char* ws = (char*)d_ws;
    size_t off = 0;
    float*          g   = (float*)(ws + off);          off += (size_t)N_TOK * NEXP * 4;
    unsigned short* xb  = (unsigned short*)(ws + off); off += (size_t)N_TOK * DIM_D * 2;
    unsigned short* w1t = (unsigned short*)(ws + off); off += (size_t)NEXP * DIM_H * DIM_D * 2;
    unsigned short* w2t = (unsigned short*)(ws + off); off += (size_t)NEXP * DIM_O * DIM_H * 2;
    unsigned short* hb  = (unsigned short*)(ws + off); off += (size_t)N_TOK * DIM_H * 2;
    (void)ws_size; (void)in_sizes; (void)n_in;

    // router gates
    gate_kernel<<<N_TOK / 256, 256, 0, stream>>>(x, Wg, bg, g);
    // activations -> bf16
    cvt_f32_bf16<<<(N_TOK * DIM_D / 4 + 255) / 256, 256, 0, stream>>>(x, xb, N_TOK * DIM_D / 4);
    // weights -> bf16, transposed so K is contiguous for the B operand
    transpose_cvt<<<dim3(DIM_H / 32, DIM_D / 32, NEXP), 256, 0, stream>>>(W1, w1t, DIM_D, DIM_H);
    transpose_cvt<<<dim3(DIM_O / 32, DIM_H / 32, NEXP), 256, 0, stream>>>(W2, w2t, DIM_H, DIM_O);
    // output accumulator
    zero_f32<<<(out_size + 255) / 256, 256, 0, stream>>>(out, out_size);

    for (int e = 0; e < NEXP; ++e) {
        // h = relu(x @ W1_e + b1_e)  -> bf16 hb [N_TOK, H]
        gemm_bf16_wmma<0><<<dim3(DIM_H / BN, N_TOK / BM), 256, 0, stream>>>(
            xb, w1t + (size_t)e * DIM_H * DIM_D, b1 + (size_t)e * DIM_H,
            nullptr, e, hb, nullptr, N_TOK, DIM_H, DIM_D);
        // out += g[:,e] * (h @ W2_e + b2_e)   (experts serialized on stream -> plain RMW)
        gemm_bf16_wmma<1><<<dim3(DIM_O / BN, N_TOK / BM), 256, 0, stream>>>(
            hb, w2t + (size_t)e * DIM_O * DIM_H, b2 + (size_t)e * DIM_O,
            g, e, nullptr, out, N_TOK, DIM_O, DIM_H);
    }
}
